// GRU_10557029614334
// MI455X (gfx1250) — compile-verified, matrix operand order checked by reading, not run on hardware
//
#include <hip/hip_runtime.h>
#include <hip/hip_bf16.h>
#include <math.h>
#include <stddef.h>

// ---------------------------------------------------------------------------
// GRU for MI455X / gfx1250 (wave32, WMMA, LDS-staged recurrence weights).
//   Phase 1: split/transpose weights to bf16 hi/lo ([N,K] row-major).
//   Phase 2: input-projection GEMMs (HBM-bound) via v_wmma_f32_16x16x32_bf16
//            with 3-term split-bf16 fp32 emulation, on-the-fly A split.
//   Phase 3: 512 sequential step kernels (graph-captured). h kept in three
//            forms (fp32 + pre-split bf16 hi/lo) so the GEMM A-operand needs
//            no conversion VALU; U/U_z slabs staged in LDS (one N-tile per
//            block, all 8 M-tiles inside) to kill the 8x L2 re-read.
// ---------------------------------------------------------------------------

typedef __attribute__((ext_vector_type(16))) __bf16 v16bf;
typedef __attribute__((ext_vector_type(8)))  __bf16 v8bf;
typedef __attribute__((ext_vector_type(8)))  float  v8f;
typedef __attribute__((ext_vector_type(4)))  float  v4f;

#define WMMA_BF16(A, B, C) \
  __builtin_amdgcn_wmma_f32_16x16x32_bf16(false, (A), false, (B), (short)0, (C), false, false)

#define BATCH 128
#define TT    512
#define DD    1024
#define UU    1024

// --- fragment loaders -------------------------------------------------------
// A fragment (fp32 source, split on the fly): lane (l&15)=row, lane>>4 selects
// K-half: groups {h*8..h*8+7} and {16+h*8..16+h*8+7}. p = lane-adjusted base.
static __device__ __forceinline__ void load_a_split(const float* __restrict__ p,
                                                    v16bf& hi, v16bf& lo) {
  const v4f* q = reinterpret_cast<const v4f*>(p);
  v4f f0 = q[0], f1 = q[1];      // K group 0
  v4f f2 = q[4], f3 = q[5];      // K group 1 (+16 floats)
  float vals[16] = {f0.x, f0.y, f0.z, f0.w, f1.x, f1.y, f1.z, f1.w,
                    f2.x, f2.y, f2.z, f2.w, f3.x, f3.y, f3.z, f3.w};
#pragma unroll
  for (int i = 0; i < 16; ++i) {
    float f  = vals[i];
    __bf16 h = (__bf16)f;
    hi[i] = h;
    lo[i] = (__bf16)(f - (float)h);
  }
}

// bf16 fragment from row-major [.,K] bf16 (global or generic): two 16B groups
// 16 elements apart in K, lane-adjusted base p; group 2 at +16 elements.
static __device__ __forceinline__ v16bf load_bf16_frag(const __bf16* __restrict__ p) {
  const v8bf* q = reinterpret_cast<const v8bf*>(p);
  v8bf b0 = q[0];
  v8bf b1 = q[2];
  v16bf r;
#pragma unroll
  for (int i = 0; i < 8; ++i) { r[i] = b0[i]; r[8 + i] = b1[i]; }
  return r;
}

// --- weight split/transpose: src [D,U] fp32 -> dst_hi/lo [U,D] bf16 ---------
__global__ void __launch_bounds__(256)
split_transpose_kernel(const float* __restrict__ src, __bf16* __restrict__ dst_hi,
                       __bf16* __restrict__ dst_lo, int D, int U) {
  int idx = blockIdx.x * blockDim.x + threadIdx.x;
  if (idx >= D * U) return;
  int u = idx / D;
  int d = idx - u * D;
  float f  = src[(size_t)d * U + u];
  __bf16 h = (__bf16)f;
  dst_hi[idx] = h;
  dst_lo[idx] = (__bf16)(f - (float)h);
}

__global__ void __launch_bounds__(256)
zero_kernel(float* __restrict__ p, int n) {
  int i = blockIdx.x * blockDim.x + threadIdx.x;
  if (i < n) p[i] = 0.0f;
}

// --- Phase 2: xw = X@W + b, xwz = X@W_z + b_z (HBM-bound) ------------------
__global__ void __launch_bounds__(256)
proj_kernel(const float* __restrict__ x,
            const __bf16* __restrict__ wt_hi, const __bf16* __restrict__ wt_lo,
            const __bf16* __restrict__ wzt_hi, const __bf16* __restrict__ wzt_lo,
            const float* __restrict__ bias, const float* __restrict__ bias_z,
            float* __restrict__ xw, float* __restrict__ xwz) {
  const int K = DD, N = UU, NT = N / 16;
  int wave = blockIdx.x * (blockDim.x >> 5) + (threadIdx.x >> 5);
  int lane = threadIdx.x & 31;
  int tileN = wave % NT;
  int tileM = wave / NT;
  int half = lane >> 4, l15 = lane & 15;

  const float*  aptr = x      + (size_t)(tileM * 16 + l15) * K + half * 8;
  const __bf16* bh   = wt_hi  + (size_t)(tileN * 16 + l15) * K + half * 8;
  const __bf16* bl   = wt_lo  + (size_t)(tileN * 16 + l15) * K + half * 8;
  const __bf16* bzh  = wzt_hi + (size_t)(tileN * 16 + l15) * K + half * 8;
  const __bf16* bzl  = wzt_lo + (size_t)(tileN * 16 + l15) * K + half * 8;

  v8f acc  = {};
  v8f accz = {};
  for (int k0 = 0; k0 < K; k0 += 32) {
    __builtin_prefetch(aptr + k0 + 128, 0, 1);
    v16bf a_hi, a_lo;
    load_a_split(aptr + k0, a_hi, a_lo);
    v16bf fbh  = load_bf16_frag(bh  + k0);
    v16bf fbl  = load_bf16_frag(bl  + k0);
    v16bf fbzh = load_bf16_frag(bzh + k0);
    v16bf fbzl = load_bf16_frag(bzl + k0);
    acc  = WMMA_BF16(a_lo, fbh,  acc);
    acc  = WMMA_BF16(a_hi, fbl,  acc);
    acc  = WMMA_BF16(a_hi, fbh,  acc);
    accz = WMMA_BF16(a_lo, fbzh, accz);
    accz = WMMA_BF16(a_hi, fbzl, accz);
    accz = WMMA_BF16(a_hi, fbzh, accz);
  }

  int n = tileN * 16 + l15;
  float bb = bias[n], bbz = bias_z[n];
#pragma unroll
  for (int r = 0; r < 8; ++r) {
    int m = tileM * 16 + half * 8 + r;
    size_t o = (size_t)m * N + n;
    xw[o]  = acc[r]  + bb;
    xwz[o] = accz[r] + bbz;
  }
}

// --- Phase 3: one recurrence step ------------------------------------------
// One block per N-tile (64 blocks); 8 waves = 8 M-tiles cover B=128.
// B-slabs (4 matrices x 16 rows x 1024 K bf16 = 128 KB) staged in LDS.
// A-operand comes pre-split (h_hi/h_lo bf16) from the previous step.
__global__ void __launch_bounds__(256)
gru_step_kernel(const float* __restrict__ h_in,
                const __bf16* __restrict__ h_hi_in, const __bf16* __restrict__ h_lo_in,
                float* __restrict__ h_out,
                __bf16* __restrict__ h_hi_out, __bf16* __restrict__ h_lo_out,
                const float* __restrict__ xw, const float* __restrict__ xwz, int t,
                const __bf16* __restrict__ ut_hi, const __bf16* __restrict__ ut_lo,
                const __bf16* __restrict__ uzt_hi, const __bf16* __restrict__ uzt_lo) {
  const int K = UU, N = UU;
  extern __shared__ __bf16 smem[];           // 4 * 16 * 1024 bf16 = 128 KB

  int tileN = blockIdx.x;                    // 0..63
  int tileM = threadIdx.x >> 5;              // wave id 0..7 == M tile
  int lane  = threadIdx.x & 31;
  int half  = lane >> 4, l15 = lane & 15;

  // --- stage the four 32KB B-slabs for this N-tile into LDS ---------------
  {
    const uint4* srcs[4] = {
        reinterpret_cast<const uint4*>(ut_hi  + (size_t)tileN * 16 * K),
        reinterpret_cast<const uint4*>(ut_lo  + (size_t)tileN * 16 * K),
        reinterpret_cast<const uint4*>(uzt_hi + (size_t)tileN * 16 * K),
        reinterpret_cast<const uint4*>(uzt_lo + (size_t)tileN * 16 * K)};
    uint4* dst = reinterpret_cast<uint4*>(smem);
#pragma unroll
    for (int m = 0; m < 4; ++m) {
      const uint4* s = srcs[m];
      for (int j = threadIdx.x; j < 2048; j += 256) dst[m * 2048 + j] = s[j];
    }
    __syncthreads();
  }

  // lane-adjusted bases (element offsets)
  int eoff = l15 * K + half * 8;
  const __bf16* sU_hi  = smem + 0 * 16 * K + eoff;
  const __bf16* sU_lo  = smem + 1 * 16 * K + eoff;
  const __bf16* sUz_hi = smem + 2 * 16 * K + eoff;
  const __bf16* sUz_lo = smem + 3 * 16 * K + eoff;

  const __bf16* ahi = h_hi_in + (size_t)(tileM * 16 + l15) * K + half * 8;
  const __bf16* alo = h_lo_in + (size_t)(tileM * 16 + l15) * K + half * 8;

  v8f acc  = {};   // h @ U   (candidate)
  v8f accz = {};   // h @ U_z (update gate)
  for (int k0 = 0; k0 < K; k0 += 32) {
    v16bf a_hi = load_bf16_frag(ahi + k0);
    v16bf a_lo = load_bf16_frag(alo + k0);
    v16bf fbh  = load_bf16_frag(sU_hi  + k0);
    v16bf fbl  = load_bf16_frag(sU_lo  + k0);
    v16bf fbzh = load_bf16_frag(sUz_hi + k0);
    v16bf fbzl = load_bf16_frag(sUz_lo + k0);
    acc  = WMMA_BF16(a_lo, fbh,  acc);
    acc  = WMMA_BF16(a_hi, fbl,  acc);
    acc  = WMMA_BF16(a_hi, fbh,  acc);
    accz = WMMA_BF16(a_lo, fbzh, accz);
    accz = WMMA_BF16(a_hi, fbzl, accz);
    accz = WMMA_BF16(a_hi, fbzh, accz);
  }

  int n = tileN * 16 + l15;
#pragma unroll
  for (int r = 0; r < 8; ++r) {
    int m = tileM * 16 + half * 8 + r;                 // batch index
    size_t ho = (size_t)m * N + n;
    size_t xo = ((size_t)m * TT + t) * (size_t)N + n;  // [B,T,U] slice
    float z    = 1.0f / (1.0f + __expf(-(xwz[xo] + accz[r])));
    float cand = tanhf(xw[xo] + acc[r]);
    float hnew = (1.0f - z) * h_in[ho] + z * cand;
    h_out[ho] = hnew;
    __bf16 hh = (__bf16)hnew;
    h_hi_out[ho] = hh;
    h_lo_out[ho] = (__bf16)(hnew - (float)hh);
  }
}

// ---------------------------------------------------------------------------
extern "C" void kernel_launch(void* const* d_in, const int* in_sizes, int n_in,
                              void* d_out, int out_size, void* d_ws, size_t ws_size,
                              hipStream_t stream) {
  (void)in_sizes; (void)n_in; (void)out_size; (void)ws_size;
  const float* inputs = (const float*)d_in[0];   // [B,T,D]
  const float* W      = (const float*)d_in[1];   // [D,U]
  const float* U      = (const float*)d_in[2];   // [U,U]
  const float* W_z    = (const float*)d_in[3];
  const float* U_z    = (const float*)d_in[4];
  // d_in[5]=W_r, d_in[6]=U_r, d_in[9]=b_r : dead code in reference
  const float* b      = (const float*)d_in[7];
  const float* b_z    = (const float*)d_in[8];

  char*  ws  = (char*)d_ws;
  size_t off = 0;
  auto carve = [&](size_t bytes) -> char* {
    char* p = ws + off;
    off = (off + bytes + 255) & ~(size_t)255;
    return p;
  };
  const size_t MKf = (size_t)BATCH * TT * UU * sizeof(float);   // 256 MB
  const size_t WB  = (size_t)UU * DD * sizeof(__bf16);          // 2 MB
  const size_t Hf  = (size_t)BATCH * UU * sizeof(float);        // 512 KB
  const size_t Hb  = (size_t)BATCH * UU * sizeof(__bf16);       // 256 KB
  float*  xw     = (float*)carve(MKf);
  float*  xwz    = (float*)carve(MKf);
  __bf16* wt_hi  = (__bf16*)carve(WB);
  __bf16* wt_lo  = (__bf16*)carve(WB);
  __bf16* wzt_hi = (__bf16*)carve(WB);
  __bf16* wzt_lo = (__bf16*)carve(WB);
  __bf16* ut_hi  = (__bf16*)carve(WB);
  __bf16* ut_lo  = (__bf16*)carve(WB);
  __bf16* uzt_hi = (__bf16*)carve(WB);
  __bf16* uzt_lo = (__bf16*)carve(WB);
  float*  hA     = (float*)carve(Hf);
  float*  hB     = (float*)carve(Hf);
  __bf16* hHiA   = (__bf16*)carve(Hb);
  __bf16* hLoA   = (__bf16*)carve(Hb);
  __bf16* hHiB   = (__bf16*)carve(Hb);
  __bf16* hLoB   = (__bf16*)carve(Hb);

  // Phase 1: weight split/transpose (4 live weight matrices)
  {
    int total = DD * UU, grid = (total + 255) / 256;
    split_transpose_kernel<<<grid, 256, 0, stream>>>(W,   wt_hi,  wt_lo,  DD, UU);
    split_transpose_kernel<<<grid, 256, 0, stream>>>(W_z, wzt_hi, wzt_lo, DD, UU);
    split_transpose_kernel<<<grid, 256, 0, stream>>>(U,   ut_hi,  ut_lo,  UU, UU);
    split_transpose_kernel<<<grid, 256, 0, stream>>>(U_z, uzt_hi, uzt_lo, UU, UU);
  }

  // h0 = 0 (fp32 + split buffers; split buffers zeroed as raw dwords)
  zero_kernel<<<(BATCH * UU + 255) / 256, 256, 0, stream>>>(hA, BATCH * UU);
  zero_kernel<<<(BATCH * UU / 2 + 255) / 256, 256, 0, stream>>>((float*)hHiA, BATCH * UU / 2);
  zero_kernel<<<(BATCH * UU / 2 + 255) / 256, 256, 0, stream>>>((float*)hLoA, BATCH * UU / 2);

  // Phase 2: input projections for all timesteps
  {
    int tiles = (BATCH * TT / 16) * (UU / 16);   // 262144
    int grid  = tiles / 8;
    proj_kernel<<<grid, 256, 0, stream>>>(inputs, wt_hi, wt_lo, wzt_hi, wzt_lo,
                                          b, b_z, xw, xwz);
  }

  // Phase 3: sequential scan; ping-pong h triple; last step writes d_out
  {
    const size_t ldsBytes = 4u * 16u * UU * sizeof(__bf16);   // 128 KB
    for (int t = 0; t < TT; ++t) {
      bool even = ((t & 1) == 0);
      const float*  hin   = even ? hA   : hB;
      const __bf16* hhi   = even ? hHiA : hHiB;
      const __bf16* hlo   = even ? hLoA : hLoB;
      float*        hout  = even ? hB   : hA;
      __bf16*       hhio  = even ? hHiB : hHiA;
      __bf16*       hloo  = even ? hLoB : hLoA;
      if (t == TT - 1) hout = (float*)d_out;
      gru_step_kernel<<<64, 256, ldsBytes, stream>>>(hin, hhi, hlo, hout, hhio, hloo,
                                                     xw, xwz, t,
                                                     ut_hi, ut_lo, uzt_hi, uzt_lo);
    }
  }
}
